// STGNN_20332375179903
// MI455X (gfx1250) — matrix operand order repaired and checked by reading, not compiled, not run on hardware
//
#include <hip/hip_runtime.h>
#include <hip/hip_bf16.h>

// ---------------- problem constants ----------------
static constexpr int NB = 8;        // batch
static constexpr int NN = 10000;    // nodes
static constexpr int NT = 12;       // timesteps
static constexpr int NE = 160000;   // edges
static constexpr int HG = 64;       // gcn hidden
static constexpr int HR = 64;       // gru hidden
static constexpr int G3 = 192;      // 3*HR
static constexpr int NP = 12;       // horizon
static constexpr int ROWS = NB * NN;        // 80000 gru rows (r = b*N + n)
static constexpr int HSTR = 72;             // padded LDS row stride (halfs)

typedef __attribute__((ext_vector_type(16))) _Float16 v16h;
typedef __attribute__((ext_vector_type(8)))  _Float16 v8h;
typedef __attribute__((ext_vector_type(8)))  float    v8f;

#define WMMA16(A, Bm, C) \
  __builtin_amdgcn_wmma_f32_16x16x32_f16(false, (A), false, (Bm), (short)0, (C), false, false)

// sigmoid via v_exp + v_rcp (no IEEE-div sequence)
static __device__ __forceinline__ v8f sigmoid8(v8f x) {
  v8f y;
#pragma unroll
  for (int j = 0; j < 8; ++j) {
    float e = __expf(-x[j]);
    y[j] = __builtin_amdgcn_rcpf(1.0f + e);
  }
  return y;
}
// tanh(x) = 2*sigmoid(2x) - 1 ; exact at +/-inf, cheap (exp + rcp + fma)
static __device__ __forceinline__ v8f tanh8(v8f x) {
  v8f y;
#pragma unroll
  for (int j = 0; j < 8; ++j) {
    float e = __expf(-2.0f * x[j]);
    y[j] = __builtin_fmaf(2.0f, __builtin_amdgcn_rcpf(1.0f + e), -1.0f);
  }
  return y;
}

// ---------------- GCN preprocessing kernels ----------------
__global__ void zero_f32(float* p, int n) {
  int i = blockIdx.x * blockDim.x + threadIdx.x;
  if (i < n) p[i] = 0.0f;
}

__global__ void deg_kernel(const int* __restrict__ ei, const float* __restrict__ ew,
                           float* __restrict__ deg) {
  int i = blockIdx.x * blockDim.x + threadIdx.x;
  if (i >= NE + NN) return;
  int d; float w;
  if (i < NE) { d = ei[NE + i]; w = ew[i]; }
  else        { d = i - NE;     w = 1.0f; }
  atomicAdd(&deg[d], w);
}

__global__ void dinv_kernel(const float* __restrict__ deg, float* __restrict__ dinv) {
  int i = blockIdx.x * blockDim.x + threadIdx.x;
  if (i >= NN) return;
  float d = deg[i];
  dinv[i] = d > 0.0f ? rsqrtf(d) : 0.0f;
}

// scatter: agg[t*ROWS + b*NN + dst] += x[b, src, t] * dinv[src]*w*dinv[dst]
__global__ void scatter_kernel(const float* __restrict__ x, const int* __restrict__ ei,
                               const float* __restrict__ ew, const float* __restrict__ dinv,
                               float* __restrict__ agg) {
  int i = blockIdx.x * blockDim.x + threadIdx.x;
  if (i >= (NE + NN) * NB * NT) return;
  int e  = i / (NB * NT);
  int bt = i % (NB * NT);
  int b  = bt / NT;
  int t  = bt % NT;
  int s, d; float w;
  if (e < NE) { s = ei[e]; d = ei[NE + e]; w = ew[e]; }
  else        { s = e - NE; d = s;         w = 1.0f; }
  float nrm = dinv[s] * w * dinv[d];
  float val = x[(b * NN + s) * NT + t] * nrm;
  atomicAdd(&agg[t * ROWS + b * NN + d], val);
}

// ---------------- fused GCN-linear + GRU + projection ----------------
// 128 threads = 4 waves; each wave owns a 16-row tile and iterates all T steps.
__global__ __launch_bounds__(128, 1)
void gru_kernel(const float* __restrict__ agg,
                const float* __restrict__ gcnW, const float* __restrict__ gcnb,
                const float* __restrict__ Wih,  const float* __restrict__ Whh,
                const float* __restrict__ bih,  const float* __restrict__ bhh,
                const float* __restrict__ Wout, const float* __restrict__ bout,
                float* __restrict__ out) {
  __shared__ _Float16 sWih[G3 * HG];       // 24 KB
  __shared__ _Float16 sWhh[G3 * HG];       // 24 KB
  __shared__ _Float16 sWout[16 * HG];      //  2 KB (rows 12..15 zero)
  __shared__ float sgW[HG], sgb[HG], sbi[G3], sbh[G3];
  __shared__ _Float16 hstage[4 * 16 * HSTR];  // per-wave transpose staging

  const int tid  = threadIdx.x;
  const int lane = tid & 31;
  const int wave = tid >> 5;

  for (int i = tid; i < G3 * HG; i += 128) {
    sWih[i] = (_Float16)Wih[i];
    sWhh[i] = (_Float16)Whh[i];
  }
  for (int i = tid; i < 16 * HG; i += 128) {
    int p = i >> 6, k = i & 63;
    sWout[i] = (p < NP) ? (_Float16)Wout[p * HG + k] : (_Float16)0.0f;
  }
  for (int i = tid; i < HG; i += 128) { sgW[i] = gcnW[i]; sgb[i] = gcnb[i]; }
  for (int i = tid; i < G3; i += 128) { sbi[i] = bih[i];  sbh[i] = bhh[i]; }
  __syncthreads();

  const int row0 = (blockIdx.x * 4 + wave) * 16;   // 5000 tiles total
  const int mrow = lane & 15;
  const int hi   = lane >> 4;     // 0: lanes 0-15, 1: lanes 16-31
  const int k0   = hi * 16;       // A-fragment K offset
  const int kbB  = hi * 16;       // B-fragment K offset
  _Float16* hst  = &hstage[wave * 16 * HSTR];

  v8f h[4];
#pragma unroll
  for (int q = 0; q < 4; ++q) h[q] = v8f{};

  v8f r[4], z[4];

  for (int t = 0; t < NT; ++t) {
    // ---- stage h (C-layout regs) -> LDS row-major f16 for A-fragment rebuild
#pragma unroll
    for (int q = 0; q < 4; ++q)
#pragma unroll
      for (int j = 0; j < 8; ++j)
        hst[(j + hi * 8) * HSTR + q * 16 + mrow] = (_Float16)h[q][j];
    asm volatile("s_wait_dscnt 0" ::: "memory");

    // ---- A_x: row m input = relu(agg_scalar * gcn_W + gcn_b), straight in f16
    float a = agg[t * ROWS + row0 + mrow];
    v16h ax[2];
#pragma unroll
    for (int kc = 0; kc < 2; ++kc) {
#pragma unroll
      for (int i = 0; i < 8; ++i) {
        int kA = kc * 32 + k0 + i;
        int kBk = kc * 32 + 8 + k0 + i;
        float va = fmaxf(a * sgW[kA] + sgb[kA], 0.0f);
        float vb = fmaxf(a * sgW[kBk] + sgb[kBk], 0.0f);
        ax[kc][i]     = (_Float16)va;
        ax[kc][i + 8] = (_Float16)vb;
      }
    }

    // ---- A_h from staged h
    v16h ah[2];
#pragma unroll
    for (int kc = 0; kc < 2; ++kc) {
      v8h lo = *reinterpret_cast<const v8h*>(&hst[mrow * HSTR + kc * 32 + k0]);
      v8h hi8 = *reinterpret_cast<const v8h*>(&hst[mrow * HSTR + kc * 32 + 8 + k0]);
      ah[kc] = __builtin_shufflevector(lo, hi8, 0,1,2,3,4,5,6,7,8,9,10,11,12,13,14,15);
    }

    // ---- 12 gate tiles: gi = x@Wih^T, gh = h@Whh^T (K=64 in two 32-chunks)
#pragma unroll
    for (int nt = 0; nt < 12; ++nt) {
      const int g = nt * 16 + mrow;                  // gate column owned by lane
      const float bi_l = sbi[g], bh_l = sbh[g];
      v8f ai = v8f{}, ag = v8f{};
#pragma unroll
      for (int kc = 0; kc < 2; ++kc) {
        v16h bI = *reinterpret_cast<const v16h*>(&sWih[g * HG + kc * 32 + kbB]);
        ai = WMMA16(ax[kc], bI, ai);
        v16h bH = *reinterpret_cast<const v16h*>(&sWhh[g * HG + kc * 32 + kbB]);
        ag = WMMA16(ah[kc], bH, ag);
      }
      if (nt < 4) {
        r[nt] = sigmoid8(ai + ag + (bi_l + bh_l));
      } else if (nt < 8) {
        z[nt - 4] = sigmoid8(ai + ag + (bi_l + bh_l));
      } else {
        const int q = nt - 8;
        v8f nn = tanh8((ai + bi_l) + r[q] * (ag + bh_l));
        // (1-z)*n + z*h == n + z*(h - n)
        h[q] = nn + z[q] * (h[q] - nn);
      }
    }
  }

  // ---- final projection: pred = h @ W_out^T + b_out (64 -> 12, padded to 16)
#pragma unroll
  for (int q = 0; q < 4; ++q)
#pragma unroll
    for (int j = 0; j < 8; ++j)
      hst[(j + hi * 8) * HSTR + q * 16 + mrow] = (_Float16)h[q][j];
  asm volatile("s_wait_dscnt 0" ::: "memory");

  v16h ahf[2];
#pragma unroll
  for (int kc = 0; kc < 2; ++kc) {
    v8h lo = *reinterpret_cast<const v8h*>(&hst[mrow * HSTR + kc * 32 + k0]);
    v8h hi8 = *reinterpret_cast<const v8h*>(&hst[mrow * HSTR + kc * 32 + 8 + k0]);
    ahf[kc] = __builtin_shufflevector(lo, hi8, 0,1,2,3,4,5,6,7,8,9,10,11,12,13,14,15);
  }
  v8f acc = v8f{};
#pragma unroll
  for (int kc = 0; kc < 2; ++kc) {
    v16h bO = *reinterpret_cast<const v16h*>(&sWout[mrow * HG + kc * 32 + kbB]);
    acc = WMMA16(ahf[kc], bO, acc);
  }
  if (mrow < NP) {
    float bo = bout[mrow];
#pragma unroll
    for (int j = 0; j < 8; ++j)
      out[(row0 + j + hi * 8) * NP + mrow] = acc[j] + bo;
  }
}

// ---------------- host launch ----------------
extern "C" void kernel_launch(void* const* d_in, const int* in_sizes, int n_in,
                              void* d_out, int out_size, void* d_ws, size_t ws_size,
                              hipStream_t stream) {
  (void)in_sizes; (void)n_in; (void)out_size; (void)ws_size;
  const float* x    = (const float*)d_in[0];
  const int*   ei   = (const int*)d_in[1];
  const float* ew   = (const float*)d_in[2];
  const float* gcnW = (const float*)d_in[3];
  const float* gcnb = (const float*)d_in[4];
  const float* Wih  = (const float*)d_in[5];
  const float* Whh  = (const float*)d_in[6];
  const float* bih  = (const float*)d_in[7];
  const float* bhh  = (const float*)d_in[8];
  const float* Wout = (const float*)d_in[9];
  const float* bout = (const float*)d_in[10];
  float* out = (float*)d_out;

  float* ws   = (float*)d_ws;
  float* deg  = ws;             // NN
  float* dinv = ws + NN;        // NN
  float* agg  = ws + 2 * NN;    // NT*ROWS = 960000

  const int zn = 2 * NN + NT * ROWS;
  zero_f32<<<(zn + 255) / 256, 256, 0, stream>>>(ws, zn);
  deg_kernel<<<((NE + NN) + 255) / 256, 256, 0, stream>>>(ei, ew, deg);
  dinv_kernel<<<(NN + 255) / 256, 256, 0, stream>>>(deg, dinv);
  const int sn = (NE + NN) * NB * NT;
  scatter_kernel<<<(sn + 255) / 256, 256, 0, stream>>>(x, ei, ew, dinv, agg);
  gru_kernel<<<(ROWS / 16) / 4, 128, 0, stream>>>(agg, gcnW, gcnb, Wih, Whh,
                                                  bih, bhh, Wout, bout, out);
}
